// Model_38654705664588
// MI455X (gfx1250) — matrix operand order repaired and checked by reading, not compiled, not run on hardware
//
#include <hip/hip_runtime.h>

#define N_USERS 17449
#define N_ITEMS 48676
#define NN (N_USERS + N_ITEMS)      // 66125 nodes
#define NPAD 66176                  // NN rounded up to 128-row GEMM blocks
#define HC 128                      // H*C
#define DIN 64
#define EDGES 2000000
#define EN (EDGES + NN)             // edges + self loops
#define NEG_SLOPE 0.2f
#define WPITCH 72                   // LDS column pitch in halfs (16B-aligned, > 64)

typedef __attribute__((ext_vector_type(16))) _Float16 v16h;
typedef __attribute__((ext_vector_type(8)))  _Float16 v8h;
typedef __attribute__((ext_vector_type(8)))  float    v8f;

// ---- order-preserving float <-> uint encoding for atomicMax segment-max ----
__device__ __forceinline__ unsigned fenc(float f) {
    unsigned u = __float_as_uint(f);
    return (u & 0x80000000u) ? ~u : (u | 0x80000000u);
}
__device__ __forceinline__ float fdec(unsigned u) {
    return __uint_as_float((u & 0x80000000u) ? (u & 0x7FFFFFFFu) : ~u);
}
__device__ __forceinline__ float lrelu(float x) { return x > 0.f ? x : x * NEG_SLOPE; }

__device__ __forceinline__ void edge_pair(const int* __restrict__ ei, int eid, int& s, int& d) {
    if (eid < EDGES) { s = ei[eid]; d = ei[EDGES + eid]; }
    else             { s = eid - EDGES; d = s; }            // self loop
}

// ============================================================================
// Kernel 1: h = x @ W via WMMA f16->f32.  One wave owns a 16-row tile
// (8 waves / block -> 128 rows).  W staged once per block into LDS as f16,
// column-major, so B fragments are contiguous ds_load_b128 pairs.
//   A (16x32 f16): lane = 16*khalf + m ; elems 0..7 -> K=8*khalf+e,
//                  elems 8..15 -> K=16+8*khalf+e   (two frags for K=64)
//   B (32x16 f16): lane = 16*khalf + n ; same K order as A
//   C (16x16 f32): lane = 16*g + n ; VGPR r -> row M = r + 8*g
// Stores are unguarded: h is padded to NPAD rows in workspace.
// ============================================================================
__global__ void __launch_bounds__(256) gemm_h_kernel(
    const float* __restrict__ users, const float* __restrict__ items,
    const float* __restrict__ W, float* __restrict__ h)
{
    __shared__ _Float16 wlds[HC * WPITCH];   // 18 KB: wlds[col*72 + k]

    // Cooperative stage of W [64 x 128] f32 -> LDS f16 column-major.
    for (int i = threadIdx.x; i < DIN * HC; i += 256) {
        int k = i >> 7, col = i & (HC - 1);
        wlds[col * WPITCH + k] = (_Float16)W[i];
    }
    __syncthreads();

    const int lane = threadIdx.x & 31;
    const int wave = threadIdx.x >> 5;
    const int row_base = blockIdx.x * 128 + wave * 16;
    const int m  = lane & 15;
    const int hi = lane >> 4;

    int row = row_base + m;
    int r = (row < NN) ? row : 0;                 // clamp loads; EXEC stays all-1s
    const float* xrow = (r < N_USERS) ? (users + (size_t)r * DIN)
                                      : (items + (size_t)(r - N_USERS) * DIN);

    // A fragments for K=0..31 (a0) and K=32..63 (a1); loaded once, reused 8x.
    v16h a0, a1;
    const int ko = hi * 8;
#pragma unroll
    for (int e = 0; e < 8; ++e) {
        a0[e]     = (_Float16)xrow[ko + e];
        a0[e + 8] = (_Float16)xrow[16 + ko + e];
        a1[e]     = (_Float16)xrow[32 + ko + e];
        a1[e + 8] = (_Float16)xrow[48 + ko + e];
    }

    const int kb = hi * 8;
#pragma unroll
    for (int ct = 0; ct < 8; ++ct) {              // 8 column tiles of 16 -> 128 outputs
        const _Float16* wc = &wlds[(ct * 16 + m) * WPITCH + kb];
        v8h l0 = *(const v8h*)(wc);               // K = kb .. kb+7
        v8h h0 = *(const v8h*)(wc + 16);          // K = 16+kb .. 16+kb+7
        v8h l1 = *(const v8h*)(wc + 32);          // K = 32+kb .. 32+kb+7
        v8h h1 = *(const v8h*)(wc + 48);          // K = 48+kb .. 48+kb+7
        v16h b0, b1;
#pragma unroll
        for (int e = 0; e < 8; ++e) {
            b0[e] = l0[e]; b0[e + 8] = h0[e];
            b1[e] = l1[e]; b1[e + 8] = h1[e];
        }
        v8f c = {};
        c = __builtin_amdgcn_wmma_f32_16x16x32_f16(false, a0, false, b0, (short)0, c, false, false);
        c = __builtin_amdgcn_wmma_f32_16x16x32_f16(false, a1, false, b1, (short)0, c, false, false);

        float* op = h + (size_t)(row_base + 8 * hi) * HC + ct * 16 + m;
#pragma unroll
        for (int rr = 0; rr < 8; ++rr)
            op[(size_t)rr * HC] = c[rr];          // branch-free: h padded to NPAD rows
    }
}

// ============================================================================
// Kernel 2: per-(node,head) attention logits  a_src/a_dst = <h[n,head,:], att>
// ============================================================================
__global__ void __launch_bounds__(256) attn_logit_kernel(
    const float* __restrict__ h,
    const float* __restrict__ att_src, const float* __restrict__ att_dst,
    float* __restrict__ a_src, float* __restrict__ a_dst)
{
    int t = blockIdx.x * blockDim.x + threadIdx.x;     // node*2 + head
    if (t >= NN * 2) return;
    int node = t >> 1, head = t & 1;
    const float* hv = h + (size_t)node * HC + head * 64;
    const float* as = att_src + head * 64;
    const float* ad = att_dst + head * 64;
    float s = 0.f, d = 0.f;
#pragma unroll 8
    for (int c = 0; c < 64; ++c) { float v = hv[c]; s += v * as[c]; d += v * ad[c]; }
    a_src[t] = s; a_dst[t] = d;
}

// ============================================================================
// Kernel 3: init out = bias (broadcast), m_u = 0 (== -inf encoded), denom = 0
// ============================================================================
__global__ void __launch_bounds__(256) init_kernel(
    float* __restrict__ out, const float* __restrict__ bias,
    unsigned* __restrict__ m_u, float* __restrict__ denom)
{
    int t = blockIdx.x * blockDim.x + threadIdx.x;
    if (t < NN * HC) out[t] = bias[t & (HC - 1)];
    if (t < NN * 2)  { m_u[t] = 0u; denom[t] = 0.f; }
}

// ============================================================================
// Kernel 4: segment max over dst (atomicMax on flip-encoded f32)
// ============================================================================
__global__ void __launch_bounds__(256) edge_max_kernel(
    const int* __restrict__ ei,
    const float* __restrict__ a_src, const float* __restrict__ a_dst,
    unsigned* __restrict__ m_u)
{
    int eid = blockIdx.x * blockDim.x + threadIdx.x;
    if (eid >= EN) return;
    int s, d; edge_pair(ei, eid, s, d);
#pragma unroll
    for (int hh = 0; hh < 2; ++hh) {
        float e = lrelu(a_src[s * 2 + hh] + a_dst[d * 2 + hh]);
        atomicMax(&m_u[d * 2 + hh], fenc(e));
    }
}

// ============================================================================
// Kernel 5: denom = segment_sum( exp(e - m[dst]) )
// ============================================================================
__global__ void __launch_bounds__(256) edge_sum_kernel(
    const int* __restrict__ ei,
    const float* __restrict__ a_src, const float* __restrict__ a_dst,
    const unsigned* __restrict__ m_u, float* __restrict__ denom)
{
    int eid = blockIdx.x * blockDim.x + threadIdx.x;
    if (eid >= EN) return;
    int s, d; edge_pair(ei, eid, s, d);
#pragma unroll
    for (int hh = 0; hh < 2; ++hh) {
        float e  = lrelu(a_src[s * 2 + hh] + a_dst[d * 2 + hh]);
        float ex = __expf(e - fdec(m_u[d * 2 + hh]));
        atomicAdd(&denom[d * 2 + hh], ex);
    }
}

// ============================================================================
// Kernel 6: scatter-add  out[dst] += alpha * h[src].  One wave per edge;
// each lane owns 4 contiguous floats (float4 gather, 4 f32 atomics).
// lane*4 in [0,128) -> head = lane>>4.  h/out are L2-resident (34 MB each).
// ============================================================================
__global__ void __launch_bounds__(256) edge_scatter_kernel(
    const int* __restrict__ ei,
    const float* __restrict__ a_src, const float* __restrict__ a_dst,
    const unsigned* __restrict__ m_u, const float* __restrict__ denom,
    const float* __restrict__ h, float* __restrict__ out)
{
    int gwave = (int)((blockIdx.x * blockDim.x + threadIdx.x) >> 5);  // edge id
    int lane  = threadIdx.x & 31;
    if (gwave >= EN) return;
    int s, d; edge_pair(ei, gwave, s, d);

    int head = lane >> 4;
    float e     = lrelu(a_src[s * 2 + head] + a_dst[d * 2 + head]);
    float mm    = fdec(m_u[d * 2 + head]);
    float alpha = __expf(e - mm) / (denom[d * 2 + head] + 1e-16f);

    const float4 hv = *(const float4*)(h + (size_t)s * HC + lane * 4);
    float* op = out + (size_t)d * HC + lane * 4;
    atomicAdd(op + 0, alpha * hv.x);
    atomicAdd(op + 1, alpha * hv.y);
    atomicAdd(op + 2, alpha * hv.z);
    atomicAdd(op + 3, alpha * hv.w);
}

extern "C" void kernel_launch(void* const* d_in, const int* in_sizes, int n_in,
                              void* d_out, int out_size, void* d_ws, size_t ws_size,
                              hipStream_t stream)
{
    const int*   ei    = (const int*)d_in[0];
    const float* users = (const float*)d_in[1];
    const float* items = (const float*)d_in[2];
    const float* W     = (const float*)d_in[3];
    const float* att_s = (const float*)d_in[4];
    const float* att_d = (const float*)d_in[5];
    const float* bias  = (const float*)d_in[6];
    float* out = (float*)d_out;

    // workspace layout (all f32/u32, 4B aligned); h padded to NPAD rows:
    float*    h     = (float*)d_ws;                       // NPAD*128
    float*    a_src = h + (size_t)NPAD * HC;              // NN*2
    float*    a_dst = a_src + (size_t)NN * 2;             // NN*2
    float*    denom = a_dst + (size_t)NN * 2;             // NN*2
    unsigned* m_u   = (unsigned*)(denom + (size_t)NN * 2);// NN*2

    dim3 blk(256);
    gemm_h_kernel      <<<dim3(NPAD / 128),                  blk, 0, stream>>>(users, items, W, h);
    attn_logit_kernel  <<<dim3((NN * 2 + 255) / 256),        blk, 0, stream>>>(h, att_s, att_d, a_src, a_dst);
    init_kernel        <<<dim3((NN * HC + 255) / 256),       blk, 0, stream>>>(out, bias, m_u, denom);
    edge_max_kernel    <<<dim3((EN + 255) / 256),            blk, 0, stream>>>(ei, a_src, a_dst, m_u);
    edge_sum_kernel    <<<dim3((EN + 255) / 256),            blk, 0, stream>>>(ei, a_src, a_dst, m_u, denom);
    edge_scatter_kernel<<<dim3(((size_t)EN * 32 + 255) / 256), blk, 0, stream>>>(ei, a_src, a_dst, m_u, denom, h, out);
}
